// LocalAttention_1726576854814
// MI455X (gfx1250) — matrix operand order repaired and checked by reading, not compiled, not run on hardware
//
#include <hip/hip_runtime.h>

typedef __attribute__((ext_vector_type(16))) _Float16 v16h;
typedef __attribute__((ext_vector_type(8)))  _Float16 v8h;
typedef __attribute__((ext_vector_type(8)))  float    v8f;

#define NTOK   49
#define HEADS  12
#define DIM    384
#define HD     32

// LDS strides (in halves / floats), chosen for bank-conflict-free 16-row access
#define XS_STRIDE 392   // X and O buffers: 64 x 384 (+8 pad)
#define QK_STRIDE 40    // per-head q/k: 64 x 32 (+8 pad)
#define VT_STRIDE 72    // v transposed: 32 x 64 (+8 pad)
#define P_STRIDE  72    // probs f16: 64 x 64 (+8 pad)
#define SS_STRIDE 65    // scores f32: 64 x 64 (+1 pad)

#define SMEM_BYTES (size_t)( (64*XS_STRIDE*2 /*Xs,Os*/ + 64*QK_STRIDE*2 /*Qs,Ks*/ \
                              + 32*VT_STRIDE + 64*P_STRIDE) * 2 + 64*SS_STRIDE*4 )

// workspace offsets (bytes)
#define WS_WQKVT 0
#define WS_WOUTT (WS_WQKVT + 3*DIM*DIM*2)          // 884736
#define WS_BQKV  (WS_WOUTT + DIM*DIM*2)            // +294912
#define WS_BM    (WS_BQKV  + 3*DIM*4)              // +4608
// BM: 64 * 12 * 49 * 49 * 4 = 7375872 bytes -> total ws ~8.56 MB

// -------- WMMA fragment loaders (ISA 7.12.2 layouts) --------
// A-matrix 16x32 f16: lane L holds row (mbase + (L&15)); halves at
// K = kbase + 8*(L>>4) + {0..7} and kbase + 16 + 8*(L>>4) + {0..7}.
__device__ __forceinline__ v16h load_a16(const _Float16* base, int stride,
                                         int mbase, int kbase, int lane) {
  int m  = mbase + (lane & 15);
  int lh = lane >> 4;
  const _Float16* p = base + m * stride + kbase + 8 * lh;
  union { v16h v; v8h h[2]; } u;
  u.h[0] = *(const v8h*)(p);
  u.h[1] = *(const v8h*)(p + 16);
  return u.v;
}

// B-matrix 32x16 f16: lane L holds col (nbase + (L&15)); halves at
// K = kbase + 8*(L>>4) + {0..7} and kbase + 16 + 8*(L>>4) + {0..7}. `base` layout: row n,
// K contiguous with `stride` halves per n.
__device__ __forceinline__ v16h load_b16(const _Float16* base, int stride,
                                         int nbase, int kbase, int lane) {
  int n  = nbase + (lane & 15);
  int lh = lane >> 4;
  const _Float16* p = base + n * stride + kbase + 8 * lh;
  union { v16h v; v8h h[2]; } u;
  u.h[0] = *(const v8h*)(p);
  u.h[1] = *(const v8h*)(p + 16);
  return u.v;
}

// -------- prep kernels --------
__global__ void prep_wqkv(const float* __restrict__ w, const float* __restrict__ bq,
                          _Float16* __restrict__ wT, float* __restrict__ bT) {
  int idx = blockIdx.x * 256 + threadIdx.x;
  const float scale = 0.17677669529663687f;  // 32^-0.5, folded into Q weights/bias
  if (idx < 3 * DIM * DIM) {
    int n = idx / DIM, k = idx - n * DIM;    // wT[n][k] = w[k][n]
    float v = w[(size_t)k * (3 * DIM) + n];
    if (n < DIM) v *= scale;
    wT[idx] = (_Float16)v;
  }
  if (idx < 3 * DIM) {
    float v = bq[idx];
    if (idx < DIM) v *= scale;
    bT[idx] = v;
  }
}

__global__ void prep_wout(const float* __restrict__ w, _Float16* __restrict__ wT) {
  int idx = blockIdx.x * 256 + threadIdx.x;
  if (idx < DIM * DIM) {
    int n = idx / DIM, k = idx - n * DIM;
    wT[idx] = (_Float16)w[(size_t)k * DIM + n];
  }
}

// Combined (rel-pos bias + window mask) table: BM[widx][h][m][n]
__global__ void prep_bm(const float* __restrict__ table, const float* __restrict__ mask,
                        float* __restrict__ BM) {
  int idx = blockIdx.x * 256 + threadIdx.x;
  const int per = HEADS * NTOK * NTOK;       // 28812
  if (idx < 64 * per) {
    int widx = idx / per;
    int rem  = idx - widx * per;
    int h    = rem / (NTOK * NTOK);
    int rem2 = rem - h * (NTOK * NTOK);
    int m = rem2 / NTOK, n = rem2 - m * NTOK;
    int i1 = m / 7, j1 = m - i1 * 7, i2 = n / 7, j2 = n - i2 * 7;
    int rel = (i1 - i2 + 6) * 13 + (j1 - j2 + 6);
    BM[idx] = table[rel * HEADS + h] + mask[(size_t)widx * NTOK * NTOK + rem2];
  }
}

// -------- fused window attention --------
__global__ void __launch_bounds__(256)
swin_attn_kernel(const float* __restrict__ x,
                 const _Float16* __restrict__ wqkvT, const float* __restrict__ bqkv,
                 const _Float16* __restrict__ woutT, const float* __restrict__ bout,
                 const float* __restrict__ BM, float* __restrict__ out) {
  extern __shared__ __align__(16) char smem_raw[];
  _Float16* Xs  = (_Float16*)smem_raw;           // 64 x XS_STRIDE
  _Float16* Os  = Xs + 64 * XS_STRIDE;           // 64 x XS_STRIDE
  _Float16* Qs  = Os + 64 * XS_STRIDE;           // 64 x QK_STRIDE
  _Float16* Ks  = Qs + 64 * QK_STRIDE;           // 64 x QK_STRIDE
  _Float16* Vt  = Ks + 64 * QK_STRIDE;           // 32 x VT_STRIDE (transposed)
  _Float16* Ps  = Vt + 32 * VT_STRIDE;           // 64 x P_STRIDE
  float*    Ssm = (float*)(Ps + 64 * P_STRIDE);  // 64 x SS_STRIDE

  const int tid  = threadIdx.x;
  const int lane = tid & 31;
  const int wave = tid >> 5;
  const int w    = blockIdx.x;        // window id, 0..2047
  const int b    = w >> 6;            // batch
  const int widx = w & 63;            // window within batch
  const int wr   = widx >> 3;
  const int wc   = widx & 7;

  const float* xw = x + (size_t)b * DIM * 3136 + (wr * 7) * 56 + wc * 7;

  // Phase 0: load window -> f16 LDS (rows 49..63 zero-padded)
  for (int idx = tid; idx < DIM * NTOK; idx += 256) {
    int c = idx / NTOK;
    int t = idx - c * NTOK;
    int i = t / 7, j = t - i * 7;
    Xs[t * XS_STRIDE + c] = (_Float16)xw[(size_t)c * 3136 + i * 56 + j];
  }
  for (int idx = tid; idx < DIM * (64 - NTOK); idx += 256) {
    int c = idx / 15;
    int t = NTOK + (idx - c * 15);
    Xs[t * XS_STRIDE + c] = (_Float16)0.f;
  }
  __syncthreads();

  for (int h = 0; h < HEADS; ++h) {
    // ---- Phase 1: q_h / k_h / v_h = X(64x384) @ W (K-steps of 32) ----
    {
      int mt = wave >> 1, nt = wave & 1;   // 4x2 output tiles per matrix
      for (int s = 0; s < 3; ++s) {        // 0=q, 1=k, 2=v
        const _Float16* wbase = wqkvT + (size_t)(s * DIM + h * HD) * DIM;
        v8f acc = {};
        for (int ks = 0; ks < 12; ++ks) {
          v16h A = load_a16(Xs, XS_STRIDE, mt * 16, ks * 32, lane);
          v16h B = load_b16(wbase, DIM, nt * 16, ks * 32, lane);
          acc = __builtin_amdgcn_wmma_f32_16x16x32_f16(false, A, false, B,
                                                       (short)0, acc, false, false);
        }
        int ncol = nt * 16 + (lane & 15);
        float bias = bqkv[s * DIM + h * HD + ncol];
        int lh = lane >> 4;
        for (int r = 0; r < 8; ++r) {
          int m = mt * 16 + r + 8 * lh;
          float v = acc[r] + bias;
          if (s == 0)      Qs[m * QK_STRIDE + ncol] = (_Float16)v;
          else if (s == 1) Ks[m * QK_STRIDE + ncol] = (_Float16)v;
          else             Vt[ncol * VT_STRIDE + m] = (_Float16)v;  // transposed
        }
      }
    }
    __syncthreads();

    // ---- Phase 2: S = q_h @ k_h^T + (bias+mask) ----
    {
      const float* bm = BM + ((size_t)widx * HEADS + h) * NTOK * NTOK;
      for (int s2 = 0; s2 < 2; ++s2) {
        int T = s2 * 8 + wave;             // 16 tiles = 4x4
        int mt = T >> 2, nt = T & 3;
        v16h A = load_a16(Qs, QK_STRIDE, mt * 16, 0, lane);
        v16h B = load_b16(Ks, QK_STRIDE, nt * 16, 0, lane);
        v8f acc = {};
        acc = __builtin_amdgcn_wmma_f32_16x16x32_f16(false, A, false, B,
                                                     (short)0, acc, false, false);
        int nn = nt * 16 + (lane & 15);
        int lh = lane >> 4;
        for (int r = 0; r < 8; ++r) {
          int m = mt * 16 + r + 8 * lh;
          if (m < NTOK && nn < NTOK)
            Ssm[m * SS_STRIDE + nn] = acc[r] + bm[m * NTOK + nn];
        }
      }
    }
    __syncthreads();

    // ---- Softmax: 4 lanes per row (adjacent lanes), 16 cols each ----
    {
      int row  = tid >> 2;
      int part = tid & 3;
      _Float16* Prow = Ps + row * P_STRIDE + part * 16;
      if (row < NTOK) {
        const float* Srow = Ssm + row * SS_STRIDE;
        int base = part * 16;
        float vbuf[16];
        float mx = -1e30f;
#pragma unroll
        for (int q = 0; q < 16; ++q) {
          int n = base + q;
          float v = (n < NTOK) ? Srow[n] : -1e30f;
          vbuf[q] = v;
          mx = fmaxf(mx, v);
        }
        mx = fmaxf(mx, __shfl_xor(mx, 1, 32));
        mx = fmaxf(mx, __shfl_xor(mx, 2, 32));
        float sum = 0.f;
#pragma unroll
        for (int q = 0; q < 16; ++q) {
          float e = __expf(vbuf[q] - mx);   // cols >= NTOK underflow to 0
          vbuf[q] = e;
          sum += e;
        }
        sum += __shfl_xor(sum, 1, 32);
        sum += __shfl_xor(sum, 2, 32);
        float inv = 1.0f / sum;
#pragma unroll
        for (int q = 0; q < 16; ++q)
          Prow[q] = (_Float16)(vbuf[q] * inv);
      } else {
#pragma unroll
        for (int q = 0; q < 16; ++q) Prow[q] = (_Float16)0.f;
      }
    }
    __syncthreads();

    // ---- Phase 3: O_h = P(64x64) @ v_h(64x32) ----
    {
      int mt = wave >> 1, nt = wave & 1;
      v8f acc = {};
      for (int ks2 = 0; ks2 < 2; ++ks2) {
        v16h A = load_a16(Ps, P_STRIDE, mt * 16, ks2 * 32, lane);
        v16h B = load_b16(Vt, VT_STRIDE, nt * 16, ks2 * 32, lane);
        acc = __builtin_amdgcn_wmma_f32_16x16x32_f16(false, A, false, B,
                                                     (short)0, acc, false, false);
      }
      int nc = nt * 16 + (lane & 15);
      int lh = lane >> 4;
      for (int r = 0; r < 8; ++r) {
        int m = mt * 16 + r + 8 * lh;
        Os[m * XS_STRIDE + h * HD + nc] = (_Float16)acc[r];
      }
    }
    __syncthreads();   // also protects Qs/Ks/Vt reuse by next head
  }

  // ---- Phase 4: Y = O(64x384) @ w_out + b_out, scatter to NCHW ----
  float* outw = out + (size_t)b * DIM * 3136 + (wr * 7) * 56 + wc * 7;
  for (int s4 = 0; s4 < 12; ++s4) {
    int T = s4 * 8 + wave;               // 96 tiles = 4m x 24n
    int mt = T & 3, nt = T >> 2;
    v8f acc = {};
    for (int ks = 0; ks < 12; ++ks) {
      v16h A = load_a16(Os, XS_STRIDE, mt * 16, ks * 32, lane);
      v16h B = load_b16(woutT, DIM, nt * 16, ks * 32, lane);
      acc = __builtin_amdgcn_wmma_f32_16x16x32_f16(false, A, false, B,
                                                   (short)0, acc, false, false);
    }
    int c = nt * 16 + (lane & 15);
    float bo = bout[c];
    int lh = lane >> 4;
    for (int r = 0; r < 8; ++r) {
      int m = mt * 16 + r + 8 * lh;
      if (m < NTOK) {
        int i = m / 7, j = m - i * 7;
        outw[(size_t)c * 3136 + i * 56 + j] = acc[r] + bo;
      }
    }
  }
}

extern "C" void kernel_launch(void* const* d_in, const int* in_sizes, int n_in,
                              void* d_out, int out_size, void* d_ws, size_t ws_size,
                              hipStream_t stream) {
  const float* x          = (const float*)d_in[0];
  const float* mask       = (const float*)d_in[1];
  const float* w_qkv      = (const float*)d_in[2];
  const float* b_qkv      = (const float*)d_in[3];
  const float* w_out      = (const float*)d_in[4];
  const float* b_out      = (const float*)d_in[5];
  const float* bias_table = (const float*)d_in[6];
  float* out = (float*)d_out;

  char* ws = (char*)d_ws;
  _Float16* wqkvT = (_Float16*)(ws + WS_WQKVT);
  _Float16* woutT = (_Float16*)(ws + WS_WOUTT);
  float*    bqkv  = (float*)(ws + WS_BQKV);
  float*    BM    = (float*)(ws + WS_BM);

  prep_wqkv<<<(3 * DIM * DIM + 255) / 256, 256, 0, stream>>>(w_qkv, b_qkv, wqkvT, bqkv);
  prep_wout<<<(DIM * DIM + 255) / 256, 256, 0, stream>>>(w_out, woutT);
  prep_bm<<<(64 * HEADS * NTOK * NTOK + 255) / 256, 256, 0, stream>>>(bias_table, mask, BM);

  (void)hipFuncSetAttribute((const void*)swin_attn_kernel,
                            hipFuncAttributeMaxDynamicSharedMemorySize,
                            (int)SMEM_BYTES);
  swin_attn_kernel<<<2048, 256, SMEM_BYTES, stream>>>(x, wqkvT, bqkv,
                                                      woutT, b_out, BM, out);
}
